// AttentionMambaHybrid_85272280695017
// MI455X (gfx1250) — compile-verified, matrix operand order checked by reading, not run on hardware
//
#include <hip/hip_runtime.h>
#include <hip/hip_bf16.h>

typedef __attribute__((ext_vector_type(16))) _Float16 v16h;
typedef __attribute__((ext_vector_type(8)))  float    v8f;
typedef __attribute__((ext_vector_type(4)))  unsigned int u32x4;
typedef __attribute__((ext_vector_type(4)))  int      i32x4;
typedef __attribute__((ext_vector_type(8)))  int      i32x8;

#define D_MODEL 256
#define D_INNER 512
#define D_STATE 16
#define D_CONV  4
#define DT_RANK 16
#define N_LAYERS_ 3
#define HD      32
#define BATCH   2
#define SEQ     2048
#define BL      (BATCH*SEQ)   // 4096 tokens

#if defined(__has_builtin)
#if __has_builtin(__builtin_amdgcn_tensor_load_to_lds)
#define HAVE_TDM 1
#endif
#endif

// ---------------------------------------------------------------------------
// Fragment loaders for V_WMMA_F32_16X16X32_F16 (wave32).
// A-operand layout (ISA 7.12.2): lane&15 = row M, lane>>4 selects K-halves
// {0..7,16..23} vs {8..15,24..31}; 16 halves per lane = 8 VGPRs.
// B-operand (32x16): lane&15 = col N, same K pattern -> identical addressing
// when memory is [N][K] row-major (our weight layout W[out][in]).
// ---------------------------------------------------------------------------
__device__ __forceinline__ v16h load_frag_rowK(const float* __restrict__ M,
                                               int ld, int lane, float scale) {
  const int hi = (lane >> 4) & 1;
  const float* p = M + (size_t)(lane & 15) * ld + hi * 8;
  v16h f;
#pragma unroll
  for (int i = 0; i < 8; ++i) f[i]     = (_Float16)(p[i]      * scale);
#pragma unroll
  for (int i = 0; i < 8; ++i) f[8 + i] = (_Float16)(p[16 + i] * scale);
  return f;
}

// B-operand when memory is [K][N] (e.g. V tile: rows = key tokens, cols = dims)
__device__ __forceinline__ v16h load_frag_colN(const float* __restrict__ M,
                                               int ld, int lane) {
  const int n = lane & 15, hi = (lane >> 4) & 1;
  v16h f;
#pragma unroll
  for (int i = 0; i < 8; ++i) f[i]     = (_Float16)M[(size_t)(hi * 8 + i) * ld + n];
#pragma unroll
  for (int i = 0; i < 8; ++i) f[8 + i] = (_Float16)M[(size_t)(16 + hi * 8 + i) * ld + n];
  return f;
}

// ---------------------------------------------------------------------------
// Tensor Data Mover: 2D tile load Global -> LDS.
// D# packing per CDNA5 ISA 8.3/8.4: group0 = {count|flags, lds_addr,
// global_addr[31:0], global_addr[56:32]|type=2}; group1 carries data_size,
// tensor_dim0/1, tile_dim0/1, tensor_dim0_stride.  Issued by one wave,
// tracked with TENSORcnt.
// ---------------------------------------------------------------------------
#ifdef HAVE_TDM
__device__ __forceinline__ void tdm_load_2d(const float* gsrc, unsigned lds_off,
                                            unsigned tile_w, unsigned tile_h,
                                            unsigned row_stride) {
  unsigned long long ga = (unsigned long long)(uintptr_t)gsrc;
  u32x4 g0;
  g0[0] = 1u;                                   // count=1, user mode
  g0[1] = lds_off;                              // lds_addr (bytes)
  g0[2] = (unsigned)(ga & 0xffffffffull);       // global_addr[31:0]
  g0[3] = (unsigned)((ga >> 32) & 0x1ffffffull) | (2u << 30); // [56:32] | type=2

  i32x8 g1;
  unsigned td0 = row_stride;                    // tensor_dim0 (no dim0 clip)
  unsigned td1 = tile_h;                        // tensor_dim1
  g1[0] = (int)(2u << 16);                      // wg_mask=0, data_size=2 (4B)
  g1[1] = (int)((td0 & 0xffffu) << 16);         // atomic_bar=0 | tdim0[15:0]
  g1[2] = (int)(((td0 >> 16) & 0xffffu) | ((td1 & 0xffffu) << 16));
  g1[3] = (int)(((td1 >> 16) & 0xffffu) | (tile_w << 16));   // | tile_dim0
  g1[4] = (int)(tile_h & 0xffffu);              // tile_dim1 ; tile_dim2=0
  g1[5] = (int)row_stride;                      // tensor_dim0_stride[31:0]
  g1[6] = 0;                                    // stride[47:32]=0
  g1[7] = 0;

  i32x4 z4 = {0, 0, 0, 0};
#if __clang_major__ >= 23
  i32x8 z8 = {0, 0, 0, 0, 0, 0, 0, 0};
  __builtin_amdgcn_tensor_load_to_lds(g0, g1, z4, z4, z8, 0);
#else
  __builtin_amdgcn_tensor_load_to_lds(g0, g1, z4, z4, 0);
#endif
}
#endif

// ---------------------------------------------------------------------------
// Block-tiled GEMM: C[M,N] = A[M,K] @ B[N,K]^T (+bias).  One 64x64 output
// tile per 256-thread block.  Per 32-wide K step, wave 0 stages A(64x32) and
// B(64x32) into LDS via TENSOR_LOAD_TO_LDS and waits on TENSORcnt; all 8
// waves then run 2 WMMAs each off the LDS tiles (16 WMMAs / 16KB staged).
// Requires M%64==0 && N%64==0 && K%32==0.
// ---------------------------------------------------------------------------
__global__ __launch_bounds__(256) void gemm_wmma_tdm(
    const float* __restrict__ A, const float* __restrict__ B,
    const float* __restrict__ bias, float* __restrict__ C,
    int M, int N, int K)
{
  __shared__ float As[64 * 32];
  __shared__ float Bs[64 * 32];

  const int lane = threadIdx.x & 31;
  const int w    = threadIdx.x >> 5;           // 0..7
  const int wm   = (w & 3) * 16;               // M sub-tile row
  const int wn   = (w >> 2) * 32;              // N sub-tile col
  const int tn   = blockIdx.x * 64;
  const int tm   = blockIdx.y * 64;
  const int hi   = (lane >> 4) & 1;
  const int l16  = lane & 15;

  v8f acc0 = {}, acc1 = {};
  for (int k0 = 0; k0 < K; k0 += 32) {
    __syncthreads();                           // prev tiles fully consumed
    if (threadIdx.x < 32) {                    // wave 0 drives the TDM
#ifdef HAVE_TDM
      tdm_load_2d(A + (size_t)tm * K + k0, (unsigned)(uintptr_t)As, 32, 64, (unsigned)K);
      tdm_load_2d(B + (size_t)tn * K + k0, (unsigned)(uintptr_t)Bs, 32, 64, (unsigned)K);
      __builtin_amdgcn_s_wait_tensorcnt(0);
#else
      for (int e = lane; e < 64 * 32; e += 32) {
        As[e] = A[(size_t)(tm + (e >> 5)) * K + k0 + (e & 31)];
        Bs[e] = B[(size_t)(tn + (e >> 5)) * K + k0 + (e & 31)];
      }
#endif
    }
    __syncthreads();

    v16h a, b0, b1;
    {
      const float* pa = As + (wm + l16) * 32 + hi * 8;
      const float* p0 = Bs + (wn + l16) * 32 + hi * 8;
      const float* p1 = Bs + (wn + 16 + l16) * 32 + hi * 8;
#pragma unroll
      for (int i = 0; i < 8; ++i) {
        a[i]      = (_Float16)pa[i];      a[8 + i]  = (_Float16)pa[16 + i];
        b0[i]     = (_Float16)p0[i];      b0[8 + i] = (_Float16)p0[16 + i];
        b1[i]     = (_Float16)p1[i];      b1[8 + i] = (_Float16)p1[16 + i];
      }
    }
    acc0 = __builtin_amdgcn_wmma_f32_16x16x32_f16(false, a, false, b0, (short)0, acc0, false, false);
    acc1 = __builtin_amdgcn_wmma_f32_16x16x32_f16(false, a, false, b1, (short)0, acc1, false, false);
  }

  const int m0 = tm + wm + hi * 8;
  const int n0 = tn + wn + l16;
  const float bv0 = bias ? bias[n0]      : 0.0f;
  const float bv1 = bias ? bias[n0 + 16] : 0.0f;
#pragma unroll
  for (int i = 0; i < 8; ++i) {
    C[(size_t)(m0 + i) * N + n0]      = acc0[i] + bv0;
    C[(size_t)(m0 + i) * N + n0 + 16] = acc1[i] + bv1;
  }
}

// ---------------------------------------------------------------------------
// Fallback GEMM (one wave per 16x16 tile) for shapes not 64-aligned (N=48).
// ---------------------------------------------------------------------------
__global__ __launch_bounds__(256) void gemm_wmma(
    const float* __restrict__ A, const float* __restrict__ B,
    const float* __restrict__ bias, float* __restrict__ C,
    int M, int N, int K)
{
  const int lane   = threadIdx.x & 31;
  const int wave   = blockIdx.x * (blockDim.x >> 5) + (threadIdx.x >> 5);
  const int tilesN = N >> 4;
  const int total  = (M >> 4) * tilesN;
  if (wave >= total) return;
  const int tm = (wave / tilesN) << 4;
  const int tn = (wave % tilesN) << 4;

  v8f acc = {};
  for (int k0 = 0; k0 < K; k0 += 32) {
    v16h a = load_frag_rowK(A + (size_t)tm * K + k0, K, lane, 1.0f);
    v16h b = load_frag_rowK(B + (size_t)tn * K + k0, K, lane, 1.0f);
    acc = __builtin_amdgcn_wmma_f32_16x16x32_f16(false, a, false, b,
                                                 (short)0, acc, false, false);
  }
  const int n  = tn + (lane & 15);
  const int m0 = tm + ((lane >> 4) & 1) * 8;
  const float bv = bias ? bias[n] : 0.0f;
#pragma unroll
  for (int i = 0; i < 8; ++i)
    C[(size_t)(m0 + i) * N + n] = acc[i] + bv;
}

// ---------------------------------------------------------------------------
// Depthwise causal conv (width 4) + SiLU.  xm = xz[:, :D_INNER].
// ---------------------------------------------------------------------------
__global__ void conv_silu(const float* __restrict__ xz,
                          const float* __restrict__ cw,
                          const float* __restrict__ cb,
                          float* __restrict__ xc)
{
  int idx = blockIdx.x * blockDim.x + threadIdx.x;
  if (idx >= BL * D_INNER) return;
  int c = idx % D_INNER;
  int t = idx / D_INNER;
  int l = t % SEQ;
  float acc = cb[c];
  const float* w = cw + c * D_CONV;
#pragma unroll
  for (int k = 0; k < D_CONV; ++k) {
    int ll = l - (D_CONV - 1) + k;
    if (ll >= 0)
      acc += xz[(size_t)(t - (D_CONV - 1) + k) * (2 * D_INNER) + c] * w[k];
  }
  xc[idx] = acc / (1.0f + __expf(-acc));   // silu
}

// ---------------------------------------------------------------------------
// dt = softplus(xdbl[:, :16] @ dt_w^T + dt_b)   (K=16: too small for WMMA)
// ---------------------------------------------------------------------------
__global__ void dt_softplus(const float* __restrict__ xdbl,
                            const float* __restrict__ W,
                            const float* __restrict__ b,
                            float* __restrict__ dt)
{
  int idx = blockIdx.x * blockDim.x + threadIdx.x;
  if (idx >= BL * D_INNER) return;
  int d = idx % D_INNER;
  int t = idx / D_INNER;
  const float* xr = xdbl + (size_t)t * 48;
  const float* w  = W + d * DT_RANK;
  float acc = b[d];
#pragma unroll
  for (int j = 0; j < DT_RANK; ++j) acc += xr[j] * w[j];
  dt[idx] = (acc > 20.0f) ? acc : __logf(1.0f + __expf(acc));
}

// ---------------------------------------------------------------------------
// Selective scan.  One lane per (batch, channel, state); 16-lane groups share
// a channel; per-timestep dot over states via shfl_xor within the group.
// ---------------------------------------------------------------------------
__global__ __launch_bounds__(256) void ssm_scan(
    const float* __restrict__ dt, const float* __restrict__ xc,
    const float* __restrict__ xdbl, const float* __restrict__ alog,
    const float* __restrict__ Dp, float* __restrict__ y)
{
  int tid = blockIdx.x * blockDim.x + threadIdx.x;   // 16384 lanes
  int s   = tid & 15;
  int ch  = tid >> 4;
  int d   = ch % D_INNER;
  int b   = ch / D_INNER;
  const float A  = -__expf(alog[d * D_STATE + s]);
  const float Dv = Dp[d];
  float h = 0.0f;
  for (int l = 0; l < SEQ; ++l) {
    size_t row = (size_t)b * SEQ + l;
    float dtv = dt[row * D_INNER + d];
    float xv  = xc[row * D_INNER + d];
    const float* bc = xdbl + row * 48;
    __builtin_prefetch(bc + 48, 0, 1);     // -> global_prefetch_b8
    float Bs = bc[16 + s];
    float Cs = bc[32 + s];
    h = __expf(dtv * A) * h + (dtv * xv) * Bs;
    float p = h * Cs;
    p += __shfl_xor(p, 1, 16);
    p += __shfl_xor(p, 2, 16);
    p += __shfl_xor(p, 4, 16);
    p += __shfl_xor(p, 8, 16);
    if (s == 0) y[row * D_INNER + d] = p + Dv * xv;
  }
}

// ---------------------------------------------------------------------------
// y = yf * silu(z),  z = xz[:, D_INNER:]
// ---------------------------------------------------------------------------
__global__ void gate_mul(const float* __restrict__ yf,
                         const float* __restrict__ xz,
                         float* __restrict__ yg)
{
  int idx = blockIdx.x * blockDim.x + threadIdx.x;
  if (idx >= BL * D_INNER) return;
  int d = idx % D_INNER;
  int t = idx / D_INNER;
  float z = xz[(size_t)t * (2 * D_INNER) + D_INNER + d];
  yg[idx] = yf[idx] * (z / (1.0f + __expf(-z)));
}

// ---------------------------------------------------------------------------
// out = LayerNorm(a + r) * g + b.  One wave per token (256 ch, 8 per lane).
// ---------------------------------------------------------------------------
__global__ __launch_bounds__(256) void ln_resid(
    const float* __restrict__ a, const float* __restrict__ r,
    const float* __restrict__ g, const float* __restrict__ b,
    float* __restrict__ out)
{
  int tok  = blockIdx.x * 8 + (threadIdx.x >> 5);
  int lane = threadIdx.x & 31;
  if (tok >= BL) return;
  const float* pa = a + (size_t)tok * D_MODEL;
  const float* pr = r + (size_t)tok * D_MODEL;
  float v[8], sum = 0.0f, sq = 0.0f;
#pragma unroll
  for (int j = 0; j < 8; ++j) {
    float x = pa[lane + 32 * j] + pr[lane + 32 * j];
    v[j] = x; sum += x; sq += x * x;
  }
#pragma unroll
  for (int off = 1; off < 32; off <<= 1) {
    sum += __shfl_xor(sum, off, 32);
    sq  += __shfl_xor(sq,  off, 32);
  }
  float mean = sum * (1.0f / D_MODEL);
  float var  = sq * (1.0f / D_MODEL) - mean * mean;
  float inv  = rsqrtf(var + 1e-5f);
  float* po = out + (size_t)tok * D_MODEL;
#pragma unroll
  for (int j = 0; j < 8; ++j) {
    int c = lane + 32 * j;
    po[c] = (v[j] - mean) * inv * g[c] + b[c];
  }
}

// ---------------------------------------------------------------------------
// Flash attention.  One wave per (b, head, 16-query tile); per 32-key tile:
// 2 WMMAs for S = Q K^T, online softmax (16-lane shfl reductions), P staged
// through LDS (C-layout -> A-layout), 2 WMMAs for P V.
// ---------------------------------------------------------------------------
__global__ __launch_bounds__(256) void flash_attn(
    const float* __restrict__ qkv, float* __restrict__ out)
{
  const int lane  = threadIdx.x & 31;
  const int wslot = threadIdx.x >> 5;
  const int wave  = blockIdx.x * 8 + wslot;    // 0..2047
  const int qt = wave & 127;
  const int hh = (wave >> 7) & 7;
  const int b  = wave >> 10;
  const int hi  = (lane >> 4) & 1;
  const int l16 = lane & 15;

  __shared__ _Float16 pl[8][16 * 32];
  _Float16* P = pl[wslot];

  const int LD = 3 * D_MODEL;
  const float scale = 0.17677669529663687f;    // 1/sqrt(32), folded into Q
  const size_t qrow = ((size_t)b * SEQ + qt * 16) * LD;
  v16h qf = load_frag_rowK(qkv + qrow + hh * HD, LD, lane, scale);

  v8f o0 = {}, o1 = {};
  float mi[8], li[8];
#pragma unroll
  for (int i = 0; i < 8; ++i) { mi[i] = -1e30f; li[i] = 0.0f; }

  for (int kt = 0; kt < SEQ / 32; ++kt) {
    const float* kbase = qkv + ((size_t)b * SEQ + kt * 32) * LD + D_MODEL + hh * HD;
    v16h k0 = load_frag_rowK(kbase, LD, lane, 1.0f);
    v16h k1 = load_frag_rowK(kbase + 16 * (size_t)LD, LD, lane, 1.0f);
    v8f z = {};
    v8f s0 = __builtin_amdgcn_wmma_f32_16x16x32_f16(false, qf, false, k0, (short)0, z, false, false);
    v8f s1 = __builtin_amdgcn_wmma_f32_16x16x32_f16(false, qf, false, k1, (short)0, z, false, false);

#pragma unroll
    for (int i = 0; i < 8; ++i) {          // row = hi*8 + i
      float r = fmaxf(s0[i], s1[i]);
      r = fmaxf(r, __shfl_xor(r, 1, 16));
      r = fmaxf(r, __shfl_xor(r, 2, 16));
      r = fmaxf(r, __shfl_xor(r, 4, 16));
      r = fmaxf(r, __shfl_xor(r, 8, 16));
      float mnew  = fmaxf(mi[i], r);
      float alpha = __expf(mi[i] - mnew);
      float p0 = __expf(s0[i] - mnew);
      float p1 = __expf(s1[i] - mnew);
      float rs = p0 + p1;
      rs += __shfl_xor(rs, 1, 16);
      rs += __shfl_xor(rs, 2, 16);
      rs += __shfl_xor(rs, 4, 16);
      rs += __shfl_xor(rs, 8, 16);
      li[i] = li[i] * alpha + rs;
      mi[i] = mnew;
      s0[i] = p0; s1[i] = p1;
      o0[i] *= alpha; o1[i] *= alpha;
    }

    __syncthreads();                        // uniform trip count across waves
#pragma unroll
    for (int i = 0; i < 8; ++i) {           // C-layout -> [query][key] in LDS
      P[(hi * 8 + i) * 32 + l16]      = (_Float16)s0[i];
      P[(hi * 8 + i) * 32 + 16 + l16] = (_Float16)s1[i];
    }
    __syncthreads();

    v16h pf;                                // reload as A-fragment
    {
      const _Float16* pr = P + l16 * 32 + hi * 8;
#pragma unroll
      for (int i = 0; i < 8; ++i) pf[i]     = pr[i];
#pragma unroll
      for (int i = 0; i < 8; ++i) pf[8 + i] = pr[16 + i];
    }
    const float* vbase = qkv + ((size_t)b * SEQ + kt * 32) * LD + 2 * D_MODEL + hh * HD;
    v16h v0 = load_frag_colN(vbase,      LD, lane);
    v16h v1 = load_frag_colN(vbase + 16, LD, lane);
    o0 = __builtin_amdgcn_wmma_f32_16x16x32_f16(false, pf, false, v0, (short)0, o0, false, false);
    o1 = __builtin_amdgcn_wmma_f32_16x16x32_f16(false, pf, false, v1, (short)0, o1, false, false);
  }

  float* orow = out + ((size_t)b * SEQ + qt * 16) * D_MODEL + hh * HD;
#pragma unroll
  for (int i = 0; i < 8; ++i) {
    float inv = 1.0f / li[i];
    orow[(size_t)(hi * 8 + i) * D_MODEL + l16]      = o0[i] * inv;
    orow[(size_t)(hi * 8 + i) * D_MODEL + 16 + l16] = o1[i] * inv;
  }
}

// ---------------------------------------------------------------------------
extern "C" void kernel_launch(void* const* d_in, const int* in_sizes, int n_in,
                              void* d_out, int out_size, void* d_ws, size_t ws_size,
                              hipStream_t stream)
{
  (void)in_sizes; (void)n_in; (void)out_size; (void)ws_size;
  const float* x       = (const float*)d_in[0];
  const float* inp_w   = (const float*)d_in[1];
  const float* inp_b   = (const float*)d_in[2];
  const float* in_proj = (const float*)d_in[3];   // [3][1024][256]
  const float* conv_w  = (const float*)d_in[4];   // [3][512][4]
  const float* conv_b  = (const float*)d_in[5];   // [3][512]
  const float* xproj_w = (const float*)d_in[6];   // [3][48][512]
  const float* dtw     = (const float*)d_in[7];   // [3][512][16]
  const float* dtb     = (const float*)d_in[8];   // [3][512]
  const float* a_log   = (const float*)d_in[9];   // [3][512][16]
  const float* d_par   = (const float*)d_in[10];  // [3][512]
  const float* out_w   = (const float*)d_in[11];  // [3][256][512]
  const float* mln_g   = (const float*)d_in[12];
  const float* mln_b   = (const float*)d_in[13];
  const float* qkv_w   = (const float*)d_in[14];  // [768][256]
  const float* qkv_b   = (const float*)d_in[15];
  const float* ao_w    = (const float*)d_in[16];  // [256][256]
  const float* ao_b    = (const float*)d_in[17];
  const float* ln_g    = (const float*)d_in[18];
  const float* ln_b    = (const float*)d_in[19];
  float* outp = (float*)d_out;

  float* ws   = (float*)d_ws;
  float* h    = ws;  ws += (size_t)BL * D_MODEL;
  float* xz   = ws;  ws += (size_t)BL * 2 * D_INNER;
  float* xc   = ws;  ws += (size_t)BL * D_INNER;
  float* xdbl = ws;  ws += (size_t)BL * 48;
  float* dt   = ws;  ws += (size_t)BL * D_INNER;
  float* yf   = ws;  ws += (size_t)BL * D_INNER;
  float* mo   = ws;  ws += (size_t)BL * D_MODEL;
  float* qkv  = ws;  ws += (size_t)BL * 3 * D_MODEL;
  float* yg   = dt;          // dt is dead after the scan
  float* ao   = xc;          // xc is dead after the mamba layers
  float* at   = xz;          // xz is dead after gating

  auto gemm = [&](const float* A, const float* B, const float* bias, float* C,
                  int M, int N, int K) {
    if ((M % 64 == 0) && (N % 64 == 0)) {
      gemm_wmma_tdm<<<dim3(N / 64, M / 64), 256, 0, stream>>>(A, B, bias, C, M, N, K);
    } else {
      int waves  = (M / 16) * (N / 16);
      int blocks = (waves + 7) / 8;
      gemm_wmma<<<blocks, 256, 0, stream>>>(A, B, bias, C, M, N, K);
    }
  };

  // h = x @ inp_w^T + inp_b
  gemm(x, inp_w, inp_b, h, BL, D_MODEL, 64);

  for (int i = 0; i < N_LAYERS_; ++i) {
    gemm(h, in_proj + (size_t)i * 2 * D_INNER * D_MODEL, nullptr, xz,
         BL, 2 * D_INNER, D_MODEL);
    conv_silu<<<(BL * D_INNER + 255) / 256, 256, 0, stream>>>(
        xz, conv_w + (size_t)i * D_INNER * D_CONV, conv_b + (size_t)i * D_INNER, xc);
    gemm(xc, xproj_w + (size_t)i * 48 * D_INNER, nullptr, xdbl, BL, 48, D_INNER);
    dt_softplus<<<(BL * D_INNER + 255) / 256, 256, 0, stream>>>(
        xdbl, dtw + (size_t)i * D_INNER * DT_RANK, dtb + (size_t)i * D_INNER, dt);
    ssm_scan<<<(BATCH * D_INNER * D_STATE) / 256, 256, 0, stream>>>(
        dt, xc, xdbl, a_log + (size_t)i * D_INNER * D_STATE,
        d_par + (size_t)i * D_INNER, yf);
    gate_mul<<<(BL * D_INNER + 255) / 256, 256, 0, stream>>>(yf, xz, yg);
    gemm(yg, out_w + (size_t)i * D_MODEL * D_INNER, nullptr, mo,
         BL, D_MODEL, D_INNER);
    ln_resid<<<(BL + 7) / 8, 256, 0, stream>>>(
        mo, h, mln_g + (size_t)i * D_MODEL, mln_b + (size_t)i * D_MODEL, h);
  }

  // MHSA
  gemm(h, qkv_w, qkv_b, qkv, BL, 3 * D_MODEL, D_MODEL);
  flash_attn<<<256, 256, 0, stream>>>(qkv, ao);
  gemm(ao, ao_w, ao_b, at, BL, D_MODEL, D_MODEL);
  ln_resid<<<(BL + 7) / 8, 256, 0, stream>>>(at, h, ln_g, ln_b, outp);
}
